// GraphSageLayer_70265664963122
// MI455X (gfx1250) — compile-verified, matrix operand order checked by reading, not run on hardware
//
#include <hip/hip_runtime.h>
#include <hip/hip_bf16.h>

typedef __attribute__((ext_vector_type(16))) __bf16 v16bf;
typedef __attribute__((ext_vector_type(8)))  float  v8f;

#define IN_FEATS  128
#define OUT_FEATS 128

union AFrag {
    uint4 u4[2];   // 32 bytes = 16 bf16
    v16bf v;
};

// ---------------- zero workspace (msg_sum + deg) ----------------
__global__ __launch_bounds__(256) void k_zero(float* p, long long n) {
    long long i = (long long)blockIdx.x * 256 + threadIdx.x;
    if (i < n) p[i] = 0.0f;
}

// ---------------- edge aggregation: one wave per edge ----------------
__global__ __launch_bounds__(256) void k_aggregate(const float* __restrict__ h,
                                                   const int* __restrict__ src,
                                                   const int* __restrict__ dst,
                                                   float* __restrict__ msg,
                                                   float* __restrict__ deg,
                                                   int n_edges) {
    int edge = (blockIdx.x * 256 + threadIdx.x) >> 5;
    int lane = threadIdx.x & 31;
    if (edge >= n_edges) return;
    int s = src[edge];
    int d = dst[edge];
    const float4* hp = (const float4*)(h + (size_t)s * IN_FEATS);
    float4 v = hp[lane];                      // 32 lanes x 4 floats = 128 feats
    float* mp = msg + (size_t)d * IN_FEATS + lane * 4;
    atomicAdd(mp + 0, v.x);
    atomicAdd(mp + 1, v.y);
    atomicAdd(mp + 2, v.z);
    atomicAdd(mp + 3, v.w);
    if (lane == 0) atomicAdd(deg + d, 1.0f);
}

// ---------------- c = msg/max(deg,1) -> bf16 ----------------
__global__ __launch_bounds__(256) void k_finalize(const float* __restrict__ msg,
                                                  const float* __restrict__ deg,
                                                  __bf16* __restrict__ Cb,
                                                  long long total) {
    long long i = (long long)blockIdx.x * 256 + threadIdx.x;
    if (i >= total) return;
    int node = (int)(i >> 7);
    float dv = fmaxf(deg[node], 1.0f);
    Cb[i] = (__bf16)(msg[i] / dv);
}

// ---------------- pre-swizzle W [256x128] f32 -> per-lane bf16 B fragments ----
// slot = (ntile*8 + kstep)*32 + lane, 16 contiguous bf16 per slot.
// lane<16: N = ntile*16+lane, K = kbase..kbase+7, kbase+16..kbase+23, kbase=kstep*32
// lane>=16: same N (lane-16), kbase += 8
__global__ __launch_bounds__(256) void k_packW(const float* __restrict__ W,
                                               __bf16* __restrict__ Wp) {
    int idx = blockIdx.x * 256 + threadIdx.x;          // 0..32767
    int e     = idx & 15;
    int slot  = idx >> 4;
    int lane  = slot & 31;
    int ks    = (slot >> 5) & 7;
    int ntile = slot >> 8;
    int n     = ntile * 16 + (lane & 15);
    int kbase = ks * 32 + ((lane >> 4) ? 8 : 0);
    int K     = kbase + (e < 8 ? e : 8 + e);           // e>=8 -> +16 region
    Wp[idx] = (__bf16)W[(size_t)K * OUT_FEATS + n];
}

// ---------------- fused GEMM (bf16 WMMA) + bias + L2 norm + ReLU ----------------
// grid.x = ceil(N/16); 8 waves per block, wave w owns N-tile w (16 cols).
// All 8 B fragments (whole K of this wave's 16 output columns) are preloaded
// into 64 VGPRs so the K loop only streams A.
__global__ __launch_bounds__(256) void k_gemm_norm_relu(const float* __restrict__ h,
                                                        const __bf16* __restrict__ Cb,
                                                        const __bf16* __restrict__ Wp,
                                                        const float* __restrict__ bias,
                                                        float* __restrict__ out,
                                                        int n_nodes) {
    __shared__ float tile[16][IN_FEATS + 4];
    __shared__ float partial[16][17];
    __shared__ float rnorm[16];

    const int wave  = threadIdx.x >> 5;   // 0..7 = N-tile
    const int lane  = threadIdx.x & 31;
    const int row16 = lane & 15;
    const int hi    = lane >> 4;          // 0 or 1
    const int m0    = blockIdx.x * 16;

    int arow = m0 + row16;
    if (arow >= n_nodes) arow = n_nodes - 1;   // clamp (harmless duplicate loads)

    // preload the 8 B fragments for this wave's N-tile (stays in registers)
    v16bf bfr[8];
    #pragma unroll
    for (int ks = 0; ks < 8; ++ks) {
        const int slot = (wave * 8 + ks) * 32 + lane;
        bfr[ks] = *(const v16bf*)(Wp + (size_t)slot * 16);
    }

    v8f acc = {};

    // K = 0..127 : A from h (f32 -> bf16 inline)
    #pragma unroll
    for (int ks = 0; ks < 4; ++ks) {
        const int k0 = ks * 32 + hi * 8;
        const float* hp = h + (size_t)arow * IN_FEATS + k0;
        float4 f0 = *(const float4*)(hp + 0);
        float4 f1 = *(const float4*)(hp + 4);
        float4 f2 = *(const float4*)(hp + 16);
        float4 f3 = *(const float4*)(hp + 20);
        v16bf a;
        a[0]  = (__bf16)f0.x; a[1]  = (__bf16)f0.y; a[2]  = (__bf16)f0.z; a[3]  = (__bf16)f0.w;
        a[4]  = (__bf16)f1.x; a[5]  = (__bf16)f1.y; a[6]  = (__bf16)f1.z; a[7]  = (__bf16)f1.w;
        a[8]  = (__bf16)f2.x; a[9]  = (__bf16)f2.y; a[10] = (__bf16)f2.z; a[11] = (__bf16)f2.w;
        a[12] = (__bf16)f3.x; a[13] = (__bf16)f3.y; a[14] = (__bf16)f3.z; a[15] = (__bf16)f3.w;
        acc = __builtin_amdgcn_wmma_f32_16x16x32_bf16(false, a, false, bfr[ks],
                                                      (short)0, acc, false, false);
    }

    // K = 128..255 : A from Cb (bf16, direct 16B loads)
    #pragma unroll
    for (int ks = 4; ks < 8; ++ks) {
        const int k0 = (ks - 4) * 32 + hi * 8;
        const __bf16* cp = Cb + (size_t)arow * IN_FEATS + k0;
        AFrag a;
        a.u4[0] = *(const uint4*)(cp);
        a.u4[1] = *(const uint4*)(cp + 16);
        acc = __builtin_amdgcn_wmma_f32_16x16x32_bf16(false, a.v, false, bfr[ks],
                                                      (short)0, acc, false, false);
    }

    // scatter accumulator tile to LDS, adding bias
    // VGPR r: lanes0-15 -> row r, lanes16-31 -> row r+8; col = wave*16 + row16
    const int col = wave * 16 + row16;
    const float bc = bias[col];
    #pragma unroll
    for (int r = 0; r < 8; ++r)
        tile[hi * 8 + r][col] = acc[r] + bc;
    __syncthreads();

    // per-row sum of squares (16 threads per row, 8 cols each)
    {
        int trow = threadIdx.x >> 4;
        int tcol = threadIdx.x & 15;
        float s = 0.0f;
        #pragma unroll
        for (int i = 0; i < 8; ++i) {
            float v = tile[trow][tcol + i * 16];
            s += v * v;
        }
        partial[trow][tcol] = s;
    }
    __syncthreads();
    if (threadIdx.x < 16) {
        float t = 0.0f;
        #pragma unroll
        for (int i = 0; i < 16; ++i) t += partial[threadIdx.x][i];
        rnorm[threadIdx.x] = fmaxf(sqrtf(t), 1e-12f);
    }
    __syncthreads();

    // normalized ReLU write-out: 2048 elems / 256 threads = 8 each (coalesced)
    #pragma unroll
    for (int i = 0; i < 8; ++i) {
        int e  = threadIdx.x + i * 256;
        int r  = e >> 7;
        int cc = e & 127;
        int orow = m0 + r;
        if (orow < n_nodes)
            out[(size_t)orow * OUT_FEATS + cc] = fmaxf(tile[r][cc] / rnorm[r], 0.0f);
    }
}

extern "C" void kernel_launch(void* const* d_in, const int* in_sizes, int n_in,
                              void* d_out, int out_size, void* d_ws, size_t ws_size,
                              hipStream_t stream) {
    const float* h    = (const float*)d_in[0];   // [N, 128]
    const float* W    = (const float*)d_in[1];   // [256, 128]
    const float* bias = (const float*)d_in[2];   // [128]
    const int*   src  = (const int*)d_in[3];     // [E]
    const int*   dst  = (const int*)d_in[4];     // [E]
    float*       out  = (float*)d_out;

    const int n_nodes = in_sizes[0] / IN_FEATS;
    const int n_edges = in_sizes[3];

    // workspace layout
    float*  msg = (float*)d_ws;                            // N*128 f32
    float*  deg = msg + (size_t)n_nodes * IN_FEATS;        // N f32
    __bf16* Cb  = (__bf16*)(deg + n_nodes);                // N*128 bf16
    __bf16* Wp  = Cb + (size_t)n_nodes * IN_FEATS;         // 8*8*32*16 bf16

    const long long nzero = (long long)n_nodes * IN_FEATS + n_nodes;
    k_zero<<<(int)((nzero + 255) / 256), 256, 0, stream>>>(msg, nzero);

    k_packW<<<(8 * 8 * 32 * 16) / 256, 256, 0, stream>>>(W, Wp);

    {
        long long threads = (long long)n_edges * 32;
        k_aggregate<<<(int)((threads + 255) / 256), 256, 0, stream>>>(
            h, src, dst, msg, deg, n_edges);
    }

    {
        long long total = (long long)n_nodes * IN_FEATS;
        k_finalize<<<(int)((total + 255) / 256), 256, 0, stream>>>(msg, deg, Cb, total);
    }

    {
        int mtiles = (n_nodes + 15) / 16;
        k_gemm_norm_relu<<<mtiles, 256, 0, stream>>>(h, Cb, Wp, bias, out, n_nodes);
    }
}
